// Attention_59450937311474
// MI455X (gfx1250) — compile-verified, matrix operand order checked by reading, not run on hardware
//
#include <hip/hip_runtime.h>
#include <hip/hip_bf16.h>

#define BB 2
#define NN 2048
#define HID 2048
#define HH 16
#define DH 128

typedef __attribute__((ext_vector_type(16))) _Float16 v16h;
typedef __attribute__((ext_vector_type(8)))  _Float16 v8h;
typedef __attribute__((ext_vector_type(8)))  float    v8f;
typedef __attribute__((ext_vector_type(4)))  float    v4f;

#if __has_builtin(__builtin_amdgcn_tensor_load_to_lds) && \
    __has_builtin(__builtin_amdgcn_s_wait_tensorcnt)
#define HAVE_TDM 1
#else
#define HAVE_TDM 0
#endif

__device__ __forceinline__ v8f wmma16(v16h a, v16h b, v8f c) {
  return __builtin_amdgcn_wmma_f32_16x16x32_f16(
      false, a, false, b, (short)0, c, false, false);
}

// A/B fragment lane pattern: element i holds K = (i<8 ? i : i+8) + koff,
// koff = (lane>=16 ? 8 : 0). Two contiguous 8-f16 (16B) blocks per lane.
__device__ __forceinline__ v16h load16_f16(const _Float16* p, int koff) {
  v8h lo = *(const v8h*)(p + koff);
  v8h hi = *(const v8h*)(p + 16 + koff);
  v16h a;
#pragma unroll
  for (int i = 0; i < 8; ++i) { a[i] = lo[i]; a[8 + i] = hi[i]; }
  return a;
}

#if HAVE_TDM
typedef __attribute__((ext_vector_type(4))) unsigned int u32x4;
typedef __attribute__((ext_vector_type(4))) int          i32x4;
typedef __attribute__((ext_vector_type(8))) int          i32x8;

// Issue one TDM 2D tile load: tile_rows rows of tile_cols f16 elements,
// row pitch = row_stride elements, global -> LDS (rows packed contiguously).
// D# layout per CDNA5 ISA (group0 128b, group1 256b).
__device__ __forceinline__ void tdm_stage_rows(unsigned lds_off,
                                               const _Float16* gsrc,
                                               unsigned tile_cols,
                                               unsigned tile_rows,
                                               unsigned row_stride) {
  unsigned long long ga = (unsigned long long)(uintptr_t)gsrc;
  u32x4 g0 = {
      1u,                                   // count=1, user descriptor
      lds_off,                              // lds_addr (bytes)
      (unsigned)ga,                         // global_addr[31:0]
      (unsigned)(ga >> 32) | (2u << 30)     // global_addr[56:32] | type=2
  };
  const unsigned td0 = row_stride;          // tensor_dim0 (elements)
  const unsigned td1 = 1u << 20;            // tensor_dim1 (rows, ample)
  i32x8 g1 = {
      (int)0x00010000u,                                  // data_size=1 (2B)
      (int)((td0 & 0xffffu) << 16),                      // tensor_dim0[15:0]
      (int)((td0 >> 16) | ((td1 & 0xffffu) << 16)),      // dim0[31:16]|dim1 lo
      (int)((td1 >> 16) | (tile_cols << 16)),            // dim1 hi | tile_dim0
      (int)(tile_rows & 0xffffu),                        // tile_dim1 (dim2=0)
      (int)row_stride,                                   // tensor_dim0_stride
      0, 0
  };
  i32x4 z = {0, 0, 0, 0};
#if defined(__clang_major__) && __clang_major__ >= 23
  i32x8 z8 = {0, 0, 0, 0, 0, 0, 0, 0};
  __builtin_amdgcn_tensor_load_to_lds(g0, g1, z, z, z8, 0);
#else
  __builtin_amdgcn_tensor_load_to_lds(g0, g1, z, z, 0);
#endif
}
#endif  // HAVE_TDM

// ---------------------------------------------------------------------------
// Kernel 0: bulk f32 -> f16 conversion (8 elems / thread, vectorized).
// ---------------------------------------------------------------------------
__global__ __launch_bounds__(256)
void cvt_kernel(const float* __restrict__ src, _Float16* __restrict__ dst,
                int n8) {
  int i = blockIdx.x * 256 + threadIdx.x;
  if (i < n8) {
    v4f a = *(const v4f*)(src + (size_t)i * 8);
    v4f b = *(const v4f*)(src + (size_t)i * 8 + 4);
    v8h o;
#pragma unroll
    for (int j = 0; j < 4; ++j) {
      o[j] = (_Float16)a[j];
      o[4 + j] = (_Float16)b[j];
    }
    *(v8h*)(dst + (size_t)i * 8) = o;
  }
}

// ---------------------------------------------------------------------------
// Kernel 1: QKV projection + RoPE.  out6144 = hidden @ w_qkv^T (both f16).
// Block = 8 waves -> 128x128 tile. B tile (128 cols x 32 K) staged in LDS,
// double-buffered via TDM (fallback: cooperative loads).
// ---------------------------------------------------------------------------
__global__ __launch_bounds__(256)
void qkv_rope_kernel(const _Float16* __restrict__ hws,
                     const float* __restrict__ cosp,
                     const float* __restrict__ sinp,
                     const _Float16* __restrict__ wqh,  // (6144,HID) f16
                     _Float16* __restrict__ qws,        // [b][h][n][d]
                     _Float16* __restrict__ kws,        // [b][h][n][d]
                     _Float16* __restrict__ vws)        // [b][h][d][n]
{
  __shared__ _Float16 bbuf[2][128 * 32];

  const int warp = threadIdx.x >> 5;
  const int lane = threadIdx.x & 31;
  const int ln   = lane & 15;
  const int koff = (lane & 16) ? 8 : 0;
  const int off  = (lane & 16) ? 8 : 0;

  const int mt = blockIdx.x / 48;
  const int ct = blockIdx.x % 48;
  const int m0 = mt * 128 + warp * 16;
  const int c0 = ct * 128;

  const _Float16* wbase = wqh + (size_t)c0 * HID;

#if HAVE_TDM
  const unsigned ldsb = (unsigned)(uintptr_t)&bbuf[0][0];
  if (warp == 0) {
    tdm_stage_rows(ldsb,        wbase + 0,  32u, 128u, HID);
    tdm_stage_rows(ldsb + 8192, wbase + 32, 32u, 128u, HID);
  }
#else
  const int sj = threadIdx.x >> 1;
  const int sp = (threadIdx.x & 1) * 16;
  {
    const _Float16* s = wbase + (size_t)sj * HID + sp;
    v8h x0 = *(const v8h*)s;
    v8h x1 = *(const v8h*)(s + 8);
    _Float16* d = bbuf[0] + sj * 32 + sp;
    *(v8h*)d = x0;
    *(v8h*)(d + 8) = x1;
  }
#endif

  const _Float16* arow = hws + (size_t)(m0 + ln) * HID;

  v8f acc[8];
#pragma unroll
  for (int s = 0; s < 8; ++s) acc[s] = (v8f)0.0f;

  for (int kb = 0; kb < HID; kb += 32) {
    const int cur = (kb >> 5) & 1;
#if HAVE_TDM
    if (warp == 0) {
      if (kb + 32 >= HID) __builtin_amdgcn_s_wait_tensorcnt(0);
      else                __builtin_amdgcn_s_wait_tensorcnt(1);
    }
    __syncthreads();   // publish bbuf[cur] to all waves
#else
    __syncthreads();
    if (kb + 32 < HID) {
      const _Float16* s = wbase + (size_t)sj * HID + kb + 32 + sp;
      v8h x0 = *(const v8h*)s;
      v8h x1 = *(const v8h*)(s + 8);
      _Float16* d = bbuf[cur ^ 1] + sj * 32 + sp;
      *(v8h*)d = x0;
      *(v8h*)(d + 8) = x1;
    }
#endif
    v16h a = load16_f16(arow + kb, koff);
#pragma unroll
    for (int s = 0; s < 8; ++s) {
      v16h b = load16_f16(bbuf[cur] + (s * 16 + ln) * 32, koff);
      acc[s] = wmma16(a, b, acc[s]);
    }
#if HAVE_TDM
    __syncthreads();   // all waves done with bbuf[cur] before DMA reuses it
    if (warp == 0 && kb + 64 < HID)
      tdm_stage_rows(ldsb + (unsigned)cur * 8192u,
                     wbase + kb + 64, 32u, 128u, HID);
#endif
  }

  const int which = c0 >> 11;                // 0=q 1=k 2=v
  const int h     = (c0 & 2047) >> 7;

#pragma unroll
  for (int r = 0; r < 8; ++r) {
    const int grow = m0 + r + off;           // flat row in B*N
    const int b = grow >> 11;
    const int n = grow & (NN - 1);
    if (which == 2) {
#pragma unroll
      for (int s = 0; s < 8; ++s) {
        const int d = s * 16 + ln;
        vws[((size_t)(b * HH + h) * DH + d) * NN + n] = (_Float16)acc[s][r];
      }
    } else {
      _Float16* dst = (which == 0) ? qws : kws;
#pragma unroll
      for (int s = 0; s < 8; ++s) {
        const int d = s * 16 + ln;
        const float cv = cosp[n * DH + d];
        const float sv = sinp[n * DH + d];
        const float rot = (s < 4) ? -acc[s + 4][r] : acc[s - 4][r];
        const float val = acc[s][r] * cv + rot * sv;
        dst[((size_t)(b * HH + h) * NN + n) * DH + d] = (_Float16)val;
      }
    }
  }
}

// ---------------------------------------------------------------------------
// Kernel 2: flash attention.  Block = (b,h, 8 query tiles); the 8 waves
// share cooperatively staged K (32x128) and V (128x32) chunks in LDS.
// ---------------------------------------------------------------------------
__global__ __launch_bounds__(256)
void attn_kernel(const _Float16* __restrict__ qws,
                 const _Float16* __restrict__ kws,
                 const _Float16* __restrict__ vws,
                 _Float16* __restrict__ aws)   // [b][n][h*DH] f16
{
  __shared__ _Float16 kbuf[32 * 128];
  __shared__ _Float16 vbuf[128 * 32];
  __shared__ _Float16 plds[8][16 * 32];

  const int warp = threadIdx.x >> 5;
  const int lane = threadIdx.x & 31;
  const int ln   = lane & 15;
  const int koff = (lane & 16) ? 8 : 0;
  const int off  = (lane & 16) ? 8 : 0;

  const int bh = blockIdx.x >> 4;            // b*H + h
  const int qg = blockIdx.x & 15;
  const int q0 = (qg * 8 + warp) * 16;

  const _Float16* qb = qws + (size_t)bh * NN * DH;
  const _Float16* kb = kws + (size_t)bh * NN * DH;
  const _Float16* vb = vws + (size_t)bh * DH * NN;

  const int kr = threadIdx.x >> 3;           // key row 0..31
  const int kp = (threadIdx.x & 7) * 16;     // 16-elem part of DH
  const int vd = threadIdx.x >> 1;           // dh 0..127
  const int vp = (threadIdx.x & 1) * 16;     // 16-elem part of keys

  v16h qa[4];
  const _Float16* qrow = qb + (size_t)(q0 + ln) * DH;
#pragma unroll
  for (int t = 0; t < 4; ++t) qa[t] = load16_f16(qrow + 32 * t, koff);

  v8f acc[8];
#pragma unroll
  for (int s = 0; s < 8; ++s) acc[s] = (v8f)0.0f;
  float mrow[8], lrow[8];
#pragma unroll
  for (int r = 0; r < 8; ++r) { mrow[r] = -1e30f; lrow[r] = 0.0f; }

  const float scale = 0.08838834764831845f;  // 1/sqrt(128)
  _Float16* pl = plds[warp];

  for (int kc = 0; kc < NN; kc += 32) {
    {
      const _Float16* s = kb + (size_t)(kc + kr) * DH + kp;
      v8h x0 = *(const v8h*)s;
      v8h x1 = *(const v8h*)(s + 8);
      *(v8h*)(kbuf + kr * 128 + kp) = x0;
      *(v8h*)(kbuf + kr * 128 + kp + 8) = x1;
      const _Float16* sv = vb + (size_t)vd * NN + kc + vp;
      v8h y0 = *(const v8h*)sv;
      v8h y1 = *(const v8h*)(sv + 8);
      *(v8h*)(vbuf + vd * 32 + vp) = y0;
      *(v8h*)(vbuf + vd * 32 + vp + 8) = y1;
    }
    __syncthreads();

    v8f s0 = (v8f)0.0f, s1 = (v8f)0.0f;
#pragma unroll
    for (int t = 0; t < 4; ++t) {
      v16h b0 = load16_f16(kbuf + ln * 128 + 32 * t, koff);
      v16h b1 = load16_f16(kbuf + (16 + ln) * 128 + 32 * t, koff);
      s0 = wmma16(qa[t], b0, s0);
      s1 = wmma16(qa[t], b1, s1);
    }

    float alpha[8];
#pragma unroll
    for (int r = 0; r < 8; ++r) {
      float mx = fmaxf(s0[r], s1[r]);
#pragma unroll
      for (int m = 1; m < 16; m <<= 1) mx = fmaxf(mx, __shfl_xor(mx, m, 32));
      mx *= scale;
      const float mnew = fmaxf(mrow[r], mx);
      alpha[r] = __expf(mrow[r] - mnew);
      const float p0 = __expf(s0[r] * scale - mnew);
      const float p1 = __expf(s1[r] * scale - mnew);
      float ps = p0 + p1;
#pragma unroll
      for (int m = 1; m < 16; m <<= 1) ps += __shfl_xor(ps, m, 32);
      lrow[r] = lrow[r] * alpha[r] + ps;
      mrow[r] = mnew;
      s0[r] = p0; s1[r] = p1;
    }

#pragma unroll
    for (int r = 0; r < 8; ++r) {
      pl[(r + off) * 32 + ln]      = (_Float16)s0[r];
      pl[(r + off) * 32 + 16 + ln] = (_Float16)s1[r];
    }
    v16h pa = load16_f16(pl + ln * 32, koff);

#pragma unroll
    for (int s = 0; s < 8; ++s)
#pragma unroll
      for (int r = 0; r < 8; ++r) acc[s][r] = acc[s][r] * alpha[r];

#pragma unroll
    for (int s = 0; s < 8; ++s) {
      v16h vfrag = load16_f16(vbuf + (s * 16 + ln) * 32, koff);
      acc[s] = wmma16(pa, vfrag, acc[s]);
    }
    __syncthreads();
  }

  const int b = bh >> 4;
  const int h = bh & 15;
#pragma unroll
  for (int r = 0; r < 8; ++r) {
    const int n = q0 + r + off;
    const float inv = 1.0f / lrow[r];
#pragma unroll
    for (int s = 0; s < 8; ++s) {
      const int d = s * 16 + ln;
      aws[((size_t)b * NN + n) * HID + h * DH + d] = (_Float16)(acc[s][r] * inv);
    }
  }
}

// ---------------------------------------------------------------------------
// Kernel 3: output projection. out = attn(B*N,HID) @ w_o(HID,HID)^T, f32 out.
// Same TDM-staged 128x128 structure as kernel 1.
// ---------------------------------------------------------------------------
__global__ __launch_bounds__(256)
void oproj_kernel(const _Float16* __restrict__ aws,
                  const _Float16* __restrict__ woh,
                  float* __restrict__ out)
{
  __shared__ _Float16 bbuf[2][128 * 32];

  const int warp = threadIdx.x >> 5;
  const int lane = threadIdx.x & 31;
  const int ln   = lane & 15;
  const int koff = (lane & 16) ? 8 : 0;
  const int off  = (lane & 16) ? 8 : 0;

  const int mt = blockIdx.x >> 4;            // 0..31
  const int ct = blockIdx.x & 15;            // 0..15
  const int m0 = mt * 128 + warp * 16;
  const int c0 = ct * 128;

  const _Float16* wbase = woh + (size_t)c0 * HID;

#if HAVE_TDM
  const unsigned ldsb = (unsigned)(uintptr_t)&bbuf[0][0];
  if (warp == 0) {
    tdm_stage_rows(ldsb,        wbase + 0,  32u, 128u, HID);
    tdm_stage_rows(ldsb + 8192, wbase + 32, 32u, 128u, HID);
  }
#else
  const int sj = threadIdx.x >> 1;
  const int sp = (threadIdx.x & 1) * 16;
  {
    const _Float16* s = wbase + (size_t)sj * HID + sp;
    v8h x0 = *(const v8h*)s;
    v8h x1 = *(const v8h*)(s + 8);
    _Float16* d = bbuf[0] + sj * 32 + sp;
    *(v8h*)d = x0;
    *(v8h*)(d + 8) = x1;
  }
#endif

  const _Float16* arow = aws + (size_t)(m0 + ln) * HID;

  v8f acc[8];
#pragma unroll
  for (int s = 0; s < 8; ++s) acc[s] = (v8f)0.0f;

  for (int kb = 0; kb < HID; kb += 32) {
    const int cur = (kb >> 5) & 1;
#if HAVE_TDM
    if (warp == 0) {
      if (kb + 32 >= HID) __builtin_amdgcn_s_wait_tensorcnt(0);
      else                __builtin_amdgcn_s_wait_tensorcnt(1);
    }
    __syncthreads();
#else
    __syncthreads();
    if (kb + 32 < HID) {
      const _Float16* s = wbase + (size_t)sj * HID + kb + 32 + sp;
      v8h x0 = *(const v8h*)s;
      v8h x1 = *(const v8h*)(s + 8);
      _Float16* d = bbuf[cur ^ 1] + sj * 32 + sp;
      *(v8h*)d = x0;
      *(v8h*)(d + 8) = x1;
    }
#endif
    v16h a = load16_f16(arow + kb, koff);
#pragma unroll
    for (int s = 0; s < 8; ++s) {
      v16h b = load16_f16(bbuf[cur] + (s * 16 + ln) * 32, koff);
      acc[s] = wmma16(a, b, acc[s]);
    }
#if HAVE_TDM
    __syncthreads();
    if (warp == 0 && kb + 64 < HID)
      tdm_stage_rows(ldsb + (unsigned)cur * 8192u,
                     wbase + kb + 64, 32u, 128u, HID);
#endif
  }

#pragma unroll
  for (int r = 0; r < 8; ++r) {
    const size_t row = (size_t)(m0 + r + off);
#pragma unroll
    for (int s = 0; s < 8; ++s)
      out[row * HID + c0 + s * 16 + ln] = acc[s][r];
  }
}

// ---------------------------------------------------------------------------
extern "C" void kernel_launch(void* const* d_in, const int* in_sizes, int n_in,
                              void* d_out, int out_size, void* d_ws, size_t ws_size,
                              hipStream_t stream) {
  const float* hidden = (const float*)d_in[0];   // (B,N,HID)
  const float* cosp   = (const float*)d_in[1];   // (N,DH)
  const float* sinp   = (const float*)d_in[2];   // (N,DH)
  const float* w_qkv  = (const float*)d_in[3];   // (3*HID,HID)
  const float* w_o    = (const float*)d_in[4];   // (HID,HID)
  float* out = (float*)d_out;

  const size_t hidElems  = (size_t)BB * NN * HID;        // 8,388,608
  const size_t wqkvElems = (size_t)3 * HID * HID;        // 12,582,912
  const size_t woElems   = (size_t)HID * HID;            // 4,194,304
  const size_t headElems = (size_t)BB * HH * NN * DH;    // 8,388,608

  _Float16* hws = (_Float16*)d_ws;
  _Float16* wqh = hws + hidElems;
  _Float16* woh = wqh + wqkvElems;
  _Float16* qws = woh + woElems;
  _Float16* kws = qws + headElems;
  _Float16* vws = kws + headElems;
  _Float16* aws = vws + headElems;

  cvt_kernel<<<(int)(hidElems / 8 / 256), 256, 0, stream>>>(hidden, hws,
                                                            (int)(hidElems / 8));
  cvt_kernel<<<(int)(wqkvElems / 8 / 256), 256, 0, stream>>>(w_qkv, wqh,
                                                             (int)(wqkvElems / 8));
  cvt_kernel<<<(int)(woElems / 8 / 256), 256, 0, stream>>>(w_o, woh,
                                                           (int)(woElems / 8));

  qkv_rope_kernel<<<1536, 256, 0, stream>>>(hws, cosp, sinp, wqh,
                                            qws, kws, vws);
  attn_kernel<<<512, 256, 0, stream>>>(qws, kws, vws, aws);
  oproj_kernel<<<512, 256, 0, stream>>>(aws, woh, out);
}
